// Attention_20083267076100
// MI455X (gfx1250) — matrix operand order repaired and checked by reading, not compiled, not run on hardware
//
#include <hip/hip_runtime.h>

// ---------------------------------------------------------------------------
// Problem constants
// ---------------------------------------------------------------------------
#define BB   4
#define NN   1024
#define DIMD 1024
#define HEADS 16
#define DH   64
#define ROTD 32
#define JTOT 1026          // NUM_NULL + N
#define JPAD 1056          // padded to multiple of 32
#define MROWS (BB * NN)    // 4096 rows for the big GEMMs

typedef __attribute__((ext_vector_type(16))) _Float16 v16h;
typedef __attribute__((ext_vector_type(8)))  float    v8f;
typedef __attribute__((ext_vector_type(2)))  _Float16 h2;

// ---------------------------------------------------------------------------
// WMMA helpers (CDNA5 layouts, CDNA5 ISA 7.12.2)
// A 16x32 f16: lane holds row m=lane&15; VGPR v holds K pair
//              k = (v>=4?16:0) + (lane>>4)*8 + (v&3)*2
// B 32x16 f16: lane holds col n=lane&15; VGPR v holds K pair
//              k = (v>=4?16:0) + (lane>>4)*8 + (v&3)*2
// C/D 16x16 f32: element (m = v + 8*(lane>>4), n = lane&15)
// ---------------------------------------------------------------------------
__device__ __forceinline__ v8f wmma_f16(v16h a, v16h b, v8f c) {
    return __builtin_amdgcn_wmma_f32_16x16x32_f16(
        false, a, false, b, (short)0, c, false, false);
}

// A_row: pointer to this lane's row (row m = lane&15) at current k offset
__device__ __forceinline__ v16h load_a_frag(const _Float16* A_row, int hi) {
    v16h a;
#pragma unroll
    for (int v = 0; v < 8; ++v) {
        int k = ((v >> 2) << 4) + (hi << 3) + ((v & 3) << 1);
        h2 p = *(const h2*)(A_row + k);
        a[2 * v]     = p[0];
        a[2 * v + 1] = p[1];
    }
    return a;
}

// Bt_row: pointer to this lane's column data (contiguous along K) at current k
__device__ __forceinline__ v16h load_b_frag(const _Float16* Bt_row, int hi) {
    v16h b;
#pragma unroll
    for (int v = 0; v < 8; ++v) {
        int k = ((v >> 2) << 4) + (hi << 3) + ((v & 3) << 1);
        h2 p = *(const h2*)(Bt_row + k);
        b[2 * v]     = p[0];
        b[2 * v + 1] = p[1];
    }
    return b;
}

// ---------------------------------------------------------------------------
// 1) LayerNorm: one block per row, f32 in -> f16 out
// ---------------------------------------------------------------------------
__global__ void ln_kernel(const float* __restrict__ x,
                          const float* __restrict__ g,
                          const float* __restrict__ bta,
                          _Float16* __restrict__ xn) {
    int row = blockIdx.x;
    const float* xr = x + (size_t)row * DIMD;
    __shared__ float sbuf[256];
    int tid = threadIdx.x;
    float s = 0.f, s2 = 0.f;
    for (int d = tid; d < DIMD; d += 256) {
        float v = xr[d];
        s += v; s2 += v * v;
    }
    sbuf[tid] = s; __syncthreads();
    for (int o = 128; o > 0; o >>= 1) { if (tid < o) sbuf[tid] += sbuf[tid + o]; __syncthreads(); }
    float mean = sbuf[0] * (1.0f / DIMD);
    __syncthreads();
    sbuf[tid] = s2; __syncthreads();
    for (int o = 128; o > 0; o >>= 1) { if (tid < o) sbuf[tid] += sbuf[tid + o]; __syncthreads(); }
    float var = sbuf[0] * (1.0f / DIMD) - mean * mean;
    float rstd = rsqrtf(var + 1e-5f);
    _Float16* xo = xn + (size_t)row * DIMD;
    for (int d = tid; d < DIMD; d += 256)
        xo[d] = (_Float16)((xr[d] - mean) * rstd * g[d] + bta[d]);
}

// ---------------------------------------------------------------------------
// 2) Weight transpose + f32->f16: W (K x N) -> Wt (N x K)
// ---------------------------------------------------------------------------
__global__ void wt_f16_kernel(const float* __restrict__ W,
                              _Float16* __restrict__ Wt, int K, int N) {
    int idx = blockIdx.x * blockDim.x + threadIdx.x;
    if (idx >= K * N) return;
    int n = idx % N;
    int k = idx / N;
    Wt[(size_t)n * K + k] = (_Float16)W[idx];
}

// ---------------------------------------------------------------------------
// 3) WMMA GEMM: C(MxN,f32) = A(MxK,f16 row-major) * Bt(NxK,f16)^T
//    one wave per 32x64 output tile: 2 A frags x 4 B frags = 8 WMMA / k-step
// ---------------------------------------------------------------------------
__global__ __launch_bounds__(32)
void gemm_wmma_kernel(const _Float16* __restrict__ A,
                      const _Float16* __restrict__ Bt,
                      float* __restrict__ C, int M, int N, int K) {
    int lane = threadIdx.x;
    int hi = lane >> 4, lo = lane & 15;
    int n0 = blockIdx.x * 64;
    int m0 = blockIdx.y * 32;
    const _Float16* arow0 = A + (size_t)(m0 + lo) * K;
    const _Float16* arow1 = A + (size_t)(m0 + 16 + lo) * K;
    const _Float16* b0r   = Bt + (size_t)(n0 + lo) * K;
    const _Float16* b1r   = Bt + (size_t)(n0 + 16 + lo) * K;
    const _Float16* b2r   = Bt + (size_t)(n0 + 32 + lo) * K;
    const _Float16* b3r   = Bt + (size_t)(n0 + 48 + lo) * K;
    v8f c00{}, c01{}, c02{}, c03{};
    v8f c10{}, c11{}, c12{}, c13{};
    for (int kk = 0; kk < K; kk += 32) {
        __builtin_prefetch(arow0 + kk + 128, 0, 1);  // global_prefetch_b8
        __builtin_prefetch(arow1 + kk + 128, 0, 1);
        v16h a0 = load_a_frag(arow0 + kk, hi);
        v16h a1 = load_a_frag(arow1 + kk, hi);
        v16h b;
        b = load_b_frag(b0r + kk, hi);
        c00 = wmma_f16(a0, b, c00);  c10 = wmma_f16(a1, b, c10);
        b = load_b_frag(b1r + kk, hi);
        c01 = wmma_f16(a0, b, c01);  c11 = wmma_f16(a1, b, c11);
        b = load_b_frag(b2r + kk, hi);
        c02 = wmma_f16(a0, b, c02);  c12 = wmma_f16(a1, b, c12);
        b = load_b_frag(b3r + kk, hi);
        c03 = wmma_f16(a0, b, c03);  c13 = wmma_f16(a1, b, c13);
    }
#pragma unroll
    for (int v = 0; v < 8; ++v) {
        int m = v + 8 * hi;
        float* crow0 = C + (size_t)(m0 + m) * N + n0;
        crow0[lo]      = c00[v];
        crow0[16 + lo] = c01[v];
        crow0[32 + lo] = c02[v];
        crow0[48 + lo] = c03[v];
        float* crow1 = C + (size_t)(m0 + 16 + m) * N + n0;
        crow1[lo]      = c10[v];
        crow1[16 + lo] = c11[v];
        crow1[32 + lo] = c12[v];
        crow1[48 + lo] = c13[v];
    }
}

// ---------------------------------------------------------------------------
// 4a) RoPE on q, pack to Q[b][h][n][64] f16
// ---------------------------------------------------------------------------
__global__ void rope_q_kernel(const float* __restrict__ q32,
                              const float* __restrict__ freqs,
                              _Float16* __restrict__ Q) {
    size_t idx = (size_t)blockIdx.x * blockDim.x + threadIdx.x; // B*N*H*64
    int d = idx & 63;
    size_t t = idx >> 6;
    int h = t & 15; t >>= 4;
    int n = (int)(t % NN);
    int b = (int)(t / NN);
    const float* qp = q32 + ((size_t)(b * NN + n) * DIMD) + h * DH;
    float val = qp[d];
    if (d < ROTD) {
        float f = freqs[n * ROTD + d];
        float rot = (d < 16) ? -qp[d + 16] : qp[d - 16];
        val = val * __cosf(f) + rot * __sinf(f);
    }
    Q[(((size_t)(b * HEADS + h) * NN) + n) * DH + d] = (_Float16)val;
}

// ---------------------------------------------------------------------------
// 4b) RoPE on k,v; pack K[b][2+n][64], Vt[b][64][2+n]
// ---------------------------------------------------------------------------
__global__ void rope_kv_kernel(const float* __restrict__ kv32,
                               const float* __restrict__ freqs,
                               _Float16* __restrict__ K,
                               _Float16* __restrict__ Vt) {
    size_t idx = (size_t)blockIdx.x * blockDim.x + threadIdx.x; // B*N*64
    int d = idx & 63;
    size_t t = idx >> 6;
    int n = (int)(t % NN);
    int b = (int)(t / NN);
    const float* kp = kv32 + (size_t)(b * NN + n) * 128;
    const float* vp = kp + 64;
    float kf = kp[d], vf = vp[d];
    if (d < ROTD) {
        float f = freqs[n * ROTD + d];
        float c = __cosf(f), s = __sinf(f);
        float krot = (d < 16) ? -kp[d + 16] : kp[d - 16];
        float vrot = (d < 16) ? -vp[d + 16] : vp[d - 16];
        kf = kf * c + krot * s;
        vf = vf * c + vrot * s;
    }
    int j = n + 2;
    K [(size_t)b * JPAD * DH + (size_t)j * DH + d] = (_Float16)kf;
    Vt[(size_t)b * DH * JPAD + (size_t)d * JPAD + j] = (_Float16)vf;
}

// ---------------------------------------------------------------------------
// 4c) null kv rows (j=0,1) + zero padding rows (j=1026..1055)
// ---------------------------------------------------------------------------
__global__ void nullpad_kernel(const float* __restrict__ null_kv,
                               _Float16* __restrict__ K,
                               _Float16* __restrict__ Vt) {
    int idx = blockIdx.x * blockDim.x + threadIdx.x; // B*32*64
    int d = idx & 63;
    int jj = (idx >> 6) & 31;
    int b = idx >> 11;
    float kf, vf;
    int j;
    if (jj < 2) {
        j = jj;
        kf = null_kv[(0 * 2 + jj) * DH + d];
        vf = null_kv[(1 * 2 + jj) * DH + d];
    } else {
        j = 1024 + jj;   // 1026..1055 (padding)
        kf = 0.f; vf = 0.f;
    }
    K [(size_t)b * JPAD * DH + (size_t)j * DH + d] = (_Float16)kf;
    Vt[(size_t)b * DH * JPAD + (size_t)d * JPAD + j] = (_Float16)vf;
}

// ---------------------------------------------------------------------------
// 5) Flash attention: one wave per (b, h, 16-row query tile)
//    S = Q K^T * scale (WMMA), online softmax in registers, O += P V (WMMA)
// ---------------------------------------------------------------------------
__global__ __launch_bounds__(32)
void attn_kernel(const _Float16* __restrict__ Q,
                 const _Float16* __restrict__ K,
                 const _Float16* __restrict__ Vt,
                 _Float16* __restrict__ O) {
    __shared__ _Float16 Pl[16 * 32];
    const float scale = 0.125f;     // DH^-0.5
    const float NEG = -3.0e38f;
    int lane = threadIdx.x;
    int hi = lane >> 4, lo = lane & 15;
    int i0 = blockIdx.x * 16;
    int h = blockIdx.y;
    int b = blockIdx.z;

    const _Float16* Qb = Q + ((size_t)(b * HEADS + h) * NN) * DH;
    const _Float16* Kb = K + (size_t)b * JPAD * DH;
    const _Float16* Vb = Vt + (size_t)b * DH * JPAD;

    const _Float16* qrow = Qb + (size_t)(i0 + lo) * DH;
    v16h qa0 = load_a_frag(qrow, hi);        // d 0..31
    v16h qa1 = load_a_frag(qrow + 32, hi);   // d 32..63

    v8f o0{}, o1{}, o2{}, o3{};
    float rmax[8], rsum[8];
#pragma unroll
    for (int v = 0; v < 8; ++v) { rmax[v] = NEG; rsum[v] = 0.f; }

    int jend = i0 + 18;                      // max visible j is i0+15+2
    for (int jc = 0; jc < jend; jc += 32) {
        // --- S tiles: keys jc..jc+15 (s0) and jc+16..jc+31 (s1) ---
        v8f s0{}, s1{};
        {
            const _Float16* krow0 = Kb + (size_t)(jc + lo) * DH;
            const _Float16* krow1 = Kb + (size_t)(jc + 16 + lo) * DH;
            v16h bk;
            bk = load_b_frag(krow0, hi);      s0 = wmma_f16(qa0, bk, s0);
            bk = load_b_frag(krow0 + 32, hi); s0 = wmma_f16(qa1, bk, s0);
            bk = load_b_frag(krow1, hi);      s1 = wmma_f16(qa0, bk, s1);
            bk = load_b_frag(krow1 + 32, hi); s1 = wmma_f16(qa1, bk, s1);
        }
        // --- online softmax (row m = v + 8*hi lives in 16-lane group) ---
#pragma unroll
        for (int v = 0; v < 8; ++v) {
            int m = v + 8 * hi;
            int i = i0 + m;
            float x0 = s0[v] * scale; if (jc + lo > i + 2)      x0 = NEG;
            float x1 = s1[v] * scale; if (jc + 16 + lo > i + 2) x1 = NEG;
            float cm = fmaxf(x0, x1);
#pragma unroll
            for (int o = 1; o < 16; o <<= 1) cm = fmaxf(cm, __shfl_xor(cm, o, 32));
            float nm = fmaxf(rmax[v], cm);
            float alpha = __expf(rmax[v] - nm);
            float p0 = __expf(x0 - nm);
            float p1 = __expf(x1 - nm);
            float ps = p0 + p1;
#pragma unroll
            for (int o = 1; o < 16; o <<= 1) ps += __shfl_xor(ps, o, 32);
            rsum[v] = rsum[v] * alpha + ps;
            rmax[v] = nm;
            o0[v] *= alpha; o1[v] *= alpha; o2[v] *= alpha; o3[v] *= alpha;
            Pl[m * 32 + lo]      = (_Float16)p0;
            Pl[m * 32 + 16 + lo] = (_Float16)p1;
        }
        __syncthreads();
        // --- O += P (16x32) * V (32x64): re-fragment P as A via LDS ---
        v16h ap = load_a_frag(Pl + lo * 32, hi);
        {
            v16h bv;
            const _Float16* v0r = Vb + (size_t)(lo)      * JPAD + jc;
            const _Float16* v1r = Vb + (size_t)(16 + lo) * JPAD + jc;
            const _Float16* v2r = Vb + (size_t)(32 + lo) * JPAD + jc;
            const _Float16* v3r = Vb + (size_t)(48 + lo) * JPAD + jc;
            bv = load_b_frag(v0r, hi); o0 = wmma_f16(ap, bv, o0);
            bv = load_b_frag(v1r, hi); o1 = wmma_f16(ap, bv, o1);
            bv = load_b_frag(v2r, hi); o2 = wmma_f16(ap, bv, o2);
            bv = load_b_frag(v3r, hi); o3 = wmma_f16(ap, bv, o3);
        }
        __syncthreads();
    }
    // --- normalize and write O[b][n][h*64+d] as f16 for final GEMM ---
#pragma unroll
    for (int v = 0; v < 8; ++v) {
        int m = v + 8 * hi;
        float inv = 1.0f / rsum[v];
        _Float16* orow = O + ((size_t)(b * NN + i0 + m) * DIMD) + h * DH;
        orow[lo]      = (_Float16)(o0[v] * inv);
        orow[16 + lo] = (_Float16)(o1[v] * inv);
        orow[32 + lo] = (_Float16)(o2[v] * inv);
        orow[48 + lo] = (_Float16)(o3[v] * inv);
    }
}

// ---------------------------------------------------------------------------
// Host-side launch
// inputs: 0=x 1=mask 2=freqs 3=ln_g 4=ln_b 5=W_q 6=W_kv 7=W_out 8=null_kv
// ---------------------------------------------------------------------------
extern "C" void kernel_launch(void* const* d_in, const int* in_sizes, int n_in,
                              void* d_out, int out_size, void* d_ws, size_t ws_size,
                              hipStream_t stream) {
    const float* x       = (const float*)d_in[0];
    const float* freqs   = (const float*)d_in[2];
    const float* ln_g    = (const float*)d_in[3];
    const float* ln_b    = (const float*)d_in[4];
    const float* W_q     = (const float*)d_in[5];
    const float* W_kv    = (const float*)d_in[6];
    const float* W_out   = (const float*)d_in[7];
    const float* null_kv = (const float*)d_in[8];
    float* out = (float*)d_out;

    char* ws = (char*)d_ws;
    auto alloc = [&](size_t bytes) {
        void* p = (void*)ws;
        ws += (bytes + 255) & ~(size_t)255;
        return p;
    };
    _Float16* xn_h   = (_Float16*)alloc((size_t)MROWS * DIMD * 2);
    _Float16* Wq_t   = (_Float16*)alloc((size_t)DIMD * DIMD * 2);
    _Float16* Wkv_t  = (_Float16*)alloc((size_t)128 * DIMD * 2);
    _Float16* Wout_t = (_Float16*)alloc((size_t)DIMD * DIMD * 2);
    float*    q32    = (float*)   alloc((size_t)MROWS * DIMD * 4);
    float*    kv32   = (float*)   alloc((size_t)MROWS * 128 * 4);
    _Float16* Q_h    = (_Float16*)alloc((size_t)BB * HEADS * NN * DH * 2);
    _Float16* K_h    = (_Float16*)alloc((size_t)BB * JPAD * DH * 2);
    _Float16* Vt_h   = (_Float16*)alloc((size_t)BB * DH * JPAD * 2);
    _Float16* O_h    = (_Float16*)alloc((size_t)MROWS * DIMD * 2);

    // 1) LayerNorm -> f16
    ln_kernel<<<MROWS, 256, 0, stream>>>(x, ln_g, ln_b, xn_h);

    // 2) weights -> transposed f16
    wt_f16_kernel<<<(DIMD * DIMD + 255) / 256, 256, 0, stream>>>(W_q, Wq_t, DIMD, DIMD);
    wt_f16_kernel<<<(DIMD * 128 + 255) / 256, 256, 0, stream>>>(W_kv, Wkv_t, DIMD, 128);
    wt_f16_kernel<<<(DIMD * DIMD + 255) / 256, 256, 0, stream>>>(W_out, Wout_t, DIMD, DIMD);

    // 3) projections (WMMA, 32x64 tiles per wave)
    gemm_wmma_kernel<<<dim3(DIMD / 64, MROWS / 32), 32, 0, stream>>>(
        xn_h, Wq_t, q32, MROWS, DIMD, DIMD);
    gemm_wmma_kernel<<<dim3(128 / 64, MROWS / 32), 32, 0, stream>>>(
        xn_h, Wkv_t, kv32, MROWS, 128, DIMD);

    // 4) RoPE + pack K / V^T + nulls/padding
    rope_q_kernel<<<(BB * NN * HEADS * DH) / 256, 256, 0, stream>>>(q32, freqs, Q_h);
    rope_kv_kernel<<<(BB * NN * DH) / 256, 256, 0, stream>>>(kv32, freqs, K_h, Vt_h);
    nullpad_kernel<<<(BB * 32 * 64) / 256, 256, 0, stream>>>(null_kv, K_h, Vt_h);

    // 5) flash attention (WMMA)
    attn_kernel<<<dim3(NN / 16, HEADS, BB), 32, 0, stream>>>(Q_h, K_h, Vt_h, O_h);

    // 6) output projection (WMMA) -> d_out
    gemm_wmma_kernel<<<dim3(DIMD / 64, MROWS / 32), 32, 0, stream>>>(
        O_h, Wout_t, out, MROWS, DIMD, DIMD);
}
